// TGNN_52913997086942
// MI455X (gfx1250) — matrix operand order repaired and checked by reading, not compiled, not run on hardware
//
#include <hip/hip_runtime.h>
#include <stdint.h>
#include <stddef.h>

typedef _Float16 half_t;
typedef _Float16 v16h __attribute__((ext_vector_type(16)));
typedef float    v8f  __attribute__((ext_vector_type(8)));

#define BM 128
#define BN 128
#define BK 32
#define LDSW 40   // halves per LDS row: 32 data + 8 pad -> rows stay 16B aligned

// ---------------------------------------------------------------------------
// WMMA fragment loads from LDS (wave32, 16x16x32 f16 layouts per CDNA5 ISA).
// A 16x32 f16: lane l -> M = l%16, khalf = l/16.
//   VGPR j (0..3): K = 2j + 8*khalf ; VGPR j (4..7): K = 16 + 2(j-4) + 8*khalf
//   => two aligned 16B ds loads at half-offsets {8*khalf, 16+8*khalf}.
// B 32x16 f16 (N-major tile): lane l -> N = l%16, khalf = l/16, same K order as A
//   => two aligned 16B ds loads at half-offsets {8*khalf, 16+8*khalf}.
// ---------------------------------------------------------------------------
__device__ __forceinline__ v16h frag_a(const half_t* As, int m, int khalf) {
  union { v16h v; uint4 q[2]; } f;
  const half_t* p = As + m * LDSW + khalf * 8;
  f.q[0] = *(const uint4*)p;
  f.q[1] = *(const uint4*)(p + 16);
  return f.v;
}
__device__ __forceinline__ v16h frag_b(const half_t* Bs, int n, int khalf) {
  union { v16h v; uint4 q[2]; } f;
  const half_t* p = Bs + n * LDSW + khalf * 8;
  f.q[0] = *(const uint4*)p;
  f.q[1] = *(const uint4*)(p + 16);
  return f.v;
}

// ---------------------------------------------------------------------------
// Fused GEMM: D[M,N](f16) = act( A[M,Kp](f16) x Bt[N,Kp](f16) + bias(f32) )
// Host guarantees: Kp % 32 == 0; A/Bt rows 16B aligned; K zero-padded to Kp
// => no K guards. OOB loader rows are CLAMPED to row 0 (their products only
// land in D rows/cols that the epilogue discards), so staging is branchless.
// 256 threads = 8 waves; block tile 128x128; wave tile 32x64 (2x4 WMMA accums).
// Double-buffered LDS with the last K-tile peeled out of the loop.
// ---------------------------------------------------------------------------
__global__ __launch_bounds__(256)
void gemm_wmma_f16(const half_t* __restrict__ A, int lda,
                   const half_t* __restrict__ Bt,
                   const float* __restrict__ bias,
                   half_t* __restrict__ D, int ldd,
                   int M, int N, int Kp, int do_relu) {
  __shared__ __align__(16) half_t As[2][BM * LDSW];
  __shared__ __align__(16) half_t Bs[2][BN * LDSW];

  const int tid    = threadIdx.x;
  const int lane   = tid & 31;
  const int wave   = tid >> 5;
  const int wr     = wave & 3;    // 4 row groups x 32 rows
  const int wc     = wave >> 2;   // 2 col groups x 64 cols
  const int lane16 = lane & 15;
  const int khalf  = lane >> 4;

  const int row0 = blockIdx.y * BM;
  const int col0 = blockIdx.x * BN;

  // Cooperative loader mapping: thread -> (row lm, 16-wide k segment lk).
  const int lm = tid >> 1;
  const int lk = (tid & 1) * 16;

  // Clamp OOB rows/cols to 0: branchless staging, results discarded in epilogue.
  const int gmL = row0 + lm;
  const int gnL = col0 + lm;
  const int gmC = (gmL < M) ? gmL : 0;
  const int gnC = (gnL < N) ? gnL : 0;
  const half_t* aRow = A  + (long long)gmC * lda + lk;
  const half_t* bRow = Bt + (long long)gnC * Kp  + lk;

  auto stage = [&](int kt, int sbuf) {
    const uint4* ap = (const uint4*)(aRow + kt);
    const uint4* bp = (const uint4*)(bRow + kt);
    uint4 a0 = ap[0], a1 = ap[1];
    uint4 b0 = bp[0], b1 = bp[1];
    __builtin_prefetch(aRow + kt + BK, 0, 3);  // global_prefetch_b8 (speculative)
    __builtin_prefetch(bRow + kt + BK, 0, 3);
    half_t* ad = &As[sbuf][lm * LDSW + lk];
    half_t* bd = &Bs[sbuf][lm * LDSW + lk];
    *(uint4*)ad       = a0;
    *(uint4*)(ad + 8) = a1;
    *(uint4*)bd       = b0;
    *(uint4*)(bd + 8) = b1;
  };

  v8f acc[2][4];
#pragma unroll
  for (int mi = 0; mi < 2; ++mi)
#pragma unroll
    for (int ni = 0; ni < 4; ++ni)
#pragma unroll
      for (int r = 0; r < 8; ++r) acc[mi][ni][r] = 0.0f;

  auto compute = [&](int sbuf) {
    const half_t* Ab = As[sbuf];
    const half_t* Bb = Bs[sbuf];
    v16h afr[2], bfr[4];
#pragma unroll
    for (int mi = 0; mi < 2; ++mi)
      afr[mi] = frag_a(Ab, wr * 32 + mi * 16 + lane16, khalf);
#pragma unroll
    for (int ni = 0; ni < 4; ++ni)
      bfr[ni] = frag_b(Bb, wc * 64 + ni * 16 + lane16, khalf);
#pragma unroll
    for (int mi = 0; mi < 2; ++mi)
#pragma unroll
      for (int ni = 0; ni < 4; ++ni)
        acc[mi][ni] = __builtin_amdgcn_wmma_f32_16x16x32_f16(
            false, afr[mi], false, bfr[ni], (short)0, acc[mi][ni], false, false);
  };

  stage(0, 0);
  __syncthreads();

  const int nk = Kp / BK;
  int buf = 0;
  for (int t = 0; t < nk - 1; ++t) {
    stage((t + 1) * BK, buf ^ 1);   // overlap next-tile staging with compute
    compute(buf);
    __syncthreads();
    buf ^= 1;
  }
  compute(buf);                      // peeled last tile: no staging, no barrier

  // Epilogue: bias + optional ReLU, f16 store.
  // C/D layout: VGPR r, lane l -> M = r + 8*(l/16), N = l%16.
#pragma unroll
  for (int mi = 0; mi < 2; ++mi) {
#pragma unroll
    for (int ni = 0; ni < 4; ++ni) {
      const int gn = col0 + wc * 64 + ni * 16 + lane16;
      if (gn >= N) continue;
      const float bv = bias ? bias[gn] : 0.0f;
      const int gmb = row0 + wr * 32 + mi * 16 + khalf * 8;
#pragma unroll
      for (int r = 0; r < 8; ++r) {
        const int gm = gmb + r;
        if (gm < M) {
          float v = acc[mi][ni][r] + bv;
          if (do_relu) v = fmaxf(v, 0.0f);
          D[(long long)gm * ldd + gn] = (half_t)v;
        }
      }
    }
  }
}

// --------------------------------------------------------------------------
// Weight conversion: W[K][N] f32 -> Wt[N][Kp] f16, zero-padded in [K,Kp).
// --------------------------------------------------------------------------
__global__ void convert_weight(const float* __restrict__ W, half_t* __restrict__ Wt,
                               int K, int N, int Kp) {
  const int i = blockIdx.x * blockDim.x + threadIdx.x;
  if (i >= N * Kp) return;
  const int n = i / Kp, k = i % Kp;
  Wt[i] = (k < K) ? (half_t)W[(long long)k * N + n] : (half_t)0.0f;
}

// Repack pair rows [NP,225] f32 -> [NP,256] f16: sections 92->96 | 92->96 | 41->64,
// at aligned column offsets 0 / 96 / 192, zero padded.
__global__ void repack_pairs(const float* __restrict__ in, half_t* __restrict__ out, int NP) {
  const int i = blockIdx.x * blockDim.x + threadIdx.x;
  if (i >= NP * 256) return;
  const int r = i / 256, c = i % 256;
  int sc; bool v;
  if (c < 96)       { sc = c;             v = (c < 92); }
  else if (c < 192) { sc = 92 + (c - 96); v = ((c - 96) < 92); }
  else              { sc = 184 + (c - 192); v = ((c - 192) < 41); }
  out[i] = v ? (half_t)in[(long long)r * 225 + sc] : (half_t)0.0f;
}

// Repack triplet rows [NT,358] f32 -> [NT,416] f16:
// 92->96 x3 at 0/96/192, 41->64 x2 at 288/352.
__global__ void repack_trip(const float* __restrict__ in, half_t* __restrict__ out, int NT) {
  const int i = blockIdx.x * blockDim.x + threadIdx.x;
  if (i >= NT * 416) return;
  const int r = i / 416, c = i % 416;
  int sc; bool v;
  if (c < 96)       { sc = c;               v = (c < 92); }
  else if (c < 192) { sc = 92 + (c - 96);   v = ((c - 96) < 92); }
  else if (c < 288) { sc = 184 + (c - 192); v = ((c - 192) < 92); }
  else if (c < 352) { sc = 276 + (c - 288); v = ((c - 288) < 41); }
  else              { sc = 317 + (c - 352); v = ((c - 352) < 41); }
  out[i] = v ? (half_t)in[(long long)r * 358 + sc] : (half_t)0.0f;
}

// Serial exclusive scan of the two small index arrays (B=512 -> negligible).
__global__ void scan2(const int* __restrict__ a, const int* __restrict__ b, int n,
                      int* __restrict__ oa, int* __restrict__ ob) {
  if (blockIdx.x == 0 && threadIdx.x == 0) {
    int s = 0;
    for (int i = 0; i < n; ++i) { oa[i] = s; s += a[i]; }
    s = 0;
    for (int i = 0; i < n; ++i) { ob[i] = s; s += b[i]; }
  }
}

// Segment mean over consecutive rows, strided f16 output (writes into h_cat).
__global__ void segment_mean(const half_t* __restrict__ X, const int* __restrict__ idx,
                             const int* __restrict__ off, half_t* __restrict__ out,
                             int C, int ldo) {
  const int g = blockIdx.x;
  const int c = blockIdx.y * blockDim.x + threadIdx.x;
  if (c >= C) return;
  const int start = off[g];
  const int cnt   = idx[g];
  const half_t* p = X + (long long)start * C + c;
  float s = 0.0f;
  for (int r = 0; r < cnt; ++r) s += (float)p[(long long)r * C];
  out[(long long)g * ldo + c] = (half_t)(s / (float)cnt);
}

// Fill h_cat cols [2048,2080): ref feature + zero padding (written every call).
__global__ void head_fill(const float* __restrict__ ref, half_t* __restrict__ hc, int Bn) {
  const int i = blockIdx.x * blockDim.x + threadIdx.x;
  if (i >= Bn * 32) return;
  const int b = i / 32, j = i % 32;
  hc[(long long)b * 2080 + 2048 + j] = (j == 0) ? (half_t)ref[b] : (half_t)0.0f;
}

// Final tiny layer: out[b][d] = h2[b][0:32] . fW3[:,d] + fb3[d]  (f32 output)
__global__ void final_layer(const half_t* __restrict__ h2, const float* __restrict__ w,
                            const float* __restrict__ bfin, float* __restrict__ out,
                            int Bn, int DT) {
  const int b = blockIdx.x * blockDim.x + threadIdx.x;
  if (b >= Bn) return;
  for (int d = 0; d < DT; ++d) {
    float s = bfin[d];
    for (int k = 0; k < 32; ++k) s += (float)h2[(long long)b * 32 + k] * w[k * DT + d];
    out[(long long)b * DT + d] = s;
  }
}

// --------------------------------------------------------------------------
static inline size_t align_up(size_t x, size_t a) { return (x + a - 1) / a * a; }

extern "C" void kernel_launch(void* const* d_in, const int* in_sizes, int n_in,
                              void* d_out, int out_size, void* d_ws, size_t ws_size,
                              hipStream_t stream) {
  if (n_in < 27) return;
  const int NAF = 92, NBF = 41;

  const float* pairs    = (const float*)d_in[0];
  const float* triplets = (const float*)d_in[1];
  const float* ref      = (const float*)d_in[2];
  const int*   idxp     = (const int*)d_in[3];
  const int*   idxt     = (const int*)d_in[4];
  const float* aW  = (const float*)d_in[5];  const float* ab  = (const float*)d_in[6];
  const float* bW  = (const float*)d_in[7];  const float* bb  = (const float*)d_in[8];
  const float* pW1 = (const float*)d_in[9];  const float* pb1 = (const float*)d_in[10];
  const float* pW2 = (const float*)d_in[11]; const float* pb2 = (const float*)d_in[12];
  const float* pW3 = (const float*)d_in[13]; const float* pb3 = (const float*)d_in[14];
  const float* tW1 = (const float*)d_in[15]; const float* tb1 = (const float*)d_in[16];
  const float* tW2 = (const float*)d_in[17]; const float* tb2 = (const float*)d_in[18];
  const float* tW3 = (const float*)d_in[19]; const float* tb3 = (const float*)d_in[20];
  const float* fW1 = (const float*)d_in[21]; const float* fb1 = (const float*)d_in[22];
  const float* fW2 = (const float*)d_in[23]; const float* fb2 = (const float*)d_in[24];
  const float* fW3 = (const float*)d_in[25]; const float* fb3 = (const float*)d_in[26];

  const int NP = in_sizes[0] / (2 * NAF + NBF);       // 225-col pair rows
  const int NT = in_sizes[1] / (3 * NAF + 2 * NBF);   // 358-col triplet rows
  const int Bn = in_sizes[2];
  const int DT = (Bn > 0) ? (out_size / Bn) : 1;

  // ---- bump allocator over d_ws ----
  char* cur = (char*)d_ws;
  auto alloc = [&](size_t bytes) -> void* {
    void* p = (void*)cur;
    cur += align_up(bytes, 256);
    return p;
  };
  (void)ws_size;

  int* offp = (int*)alloc((size_t)Bn * 4);
  int* offt = (int*)alloc((size_t)Bn * 4);

  half_t* aWt  = (half_t*)alloc((size_t)128 * 96 * 2);
  half_t* bWt  = (half_t*)alloc((size_t)128 * 64 * 2);
  half_t* pW1t = (half_t*)alloc((size_t)512 * 384 * 2);
  half_t* pW2t = (half_t*)alloc((size_t)256 * 512 * 2);
  half_t* pW3t = (half_t*)alloc((size_t)1024 * 256 * 2);
  half_t* tW1t = (half_t*)alloc((size_t)1024 * 640 * 2);
  half_t* tW2t = (half_t*)alloc((size_t)256 * 1024 * 2);
  half_t* tW3t = (half_t*)alloc((size_t)1024 * 256 * 2);
  half_t* fW1t = (half_t*)alloc((size_t)512 * 2080 * 2);
  half_t* fW2t = (half_t*)alloc((size_t)32 * 512 * 2);

  half_t* h_cat = (half_t*)alloc((size_t)Bn * 2080 * 2);
  half_t* h1    = (half_t*)alloc((size_t)Bn * 512 * 2);
  half_t* h2    = (half_t*)alloc((size_t)Bn * 32 * 2);

  half_t* packP = (half_t*)alloc((size_t)NP * 256 * 2);
  half_t* packT = (half_t*)alloc((size_t)NT * 416 * 2);

  size_t bufA_elems = (size_t)NP * 384;
  if ((size_t)NT * 640 > bufA_elems) bufA_elems = (size_t)NT * 640;
  size_t bufB_elems = (size_t)((NP > NT) ? NP : NT) * 1024;
  half_t* bufA = (half_t*)alloc(bufA_elems * 2);
  half_t* bufB = (half_t*)alloc(bufB_elems * 2);

  auto gemm = [&](const half_t* A, int lda, const half_t* Bt, const float* bias,
                  half_t* D, int ldd, int M, int N, int Kp, int relu) {
    dim3 grid((N + BN - 1) / BN, (M + BM - 1) / BM);
    gemm_wmma_f16<<<grid, 256, 0, stream>>>(A, lda, Bt, bias, D, ldd, M, N, Kp, relu);
  };
  auto conv = [&](const float* W, half_t* Wt, int K, int N, int Kp) {
    const int total = N * Kp;
    convert_weight<<<(total + 255) / 256, 256, 0, stream>>>(W, Wt, K, N, Kp);
  };

  // ---- prep: scans, weight conversion, input repacks ----
  scan2<<<1, 1, 0, stream>>>(idxp, idxt, Bn, offp, offt);
  conv(aW,  aWt,  92,   128, 96);
  conv(bW,  bWt,  41,   128, 64);
  conv(pW1, pW1t, 384,  512, 384);
  conv(pW2, pW2t, 512,  256, 512);
  conv(pW3, pW3t, 256, 1024, 256);
  conv(tW1, tW1t, 640, 1024, 640);
  conv(tW2, tW2t, 1024, 256, 1024);
  conv(tW3, tW3t, 256, 1024, 256);
  conv(fW1, fW1t, 2049, 512, 2080);
  conv(fW2, fW2t, 512,   32, 512);
  repack_pairs<<<((NP * 256) + 255) / 256, 256, 0, stream>>>(pairs, packP, NP);
  repack_trip<<<((NT * 416) + 255) / 256, 256, 0, stream>>>(triplets, packT, NT);

  // ---- pair path ----
  half_t* X0p = bufA;  // [NP,384]
  gemm(packP + 0,   256, aWt, ab, X0p + 0,   384, NP, 128, 96, 1);
  gemm(packP + 96,  256, aWt, ab, X0p + 128, 384, NP, 128, 96, 1);
  gemm(packP + 192, 256, bWt, bb, X0p + 256, 384, NP, 128, 64, 1);
  half_t* X1p = bufB;  // [NP,512]
  gemm(X0p, 384, pW1t, pb1, X1p, 512, NP, 512, 384, 1);
  half_t* X2p = bufA;  // [NP,256]
  gemm(X1p, 512, pW2t, pb2, X2p, 256, NP, 256, 512, 1);
  half_t* X3p = bufB;  // [NP,1024]
  gemm(X2p, 256, pW3t, pb3, X3p, 1024, NP, 1024, 256, 0);
  {
    dim3 g(Bn, (1024 + 255) / 256);
    segment_mean<<<g, 256, 0, stream>>>(X3p, idxp, offp, h_cat + 0, 1024, 2080);
  }

  // ---- triplet path ----
  half_t* X0t = bufA;  // [NT,640]
  gemm(packT + 0,   416, aWt, ab, X0t + 0,   640, NT, 128, 96, 1);
  gemm(packT + 96,  416, aWt, ab, X0t + 128, 640, NT, 128, 96, 1);
  gemm(packT + 192, 416, aWt, ab, X0t + 256, 640, NT, 128, 96, 1);
  gemm(packT + 288, 416, bWt, bb, X0t + 384, 640, NT, 128, 64, 1);
  gemm(packT + 352, 416, bWt, bb, X0t + 512, 640, NT, 128, 64, 1);
  half_t* X1t = bufB;  // [NT,1024]
  gemm(X0t, 640, tW1t, tb1, X1t, 1024, NT, 1024, 640, 1);
  half_t* X2t = bufA;  // [NT,256]
  gemm(X1t, 1024, tW2t, tb2, X2t, 256, NT, 256, 1024, 1);
  half_t* X3t = bufB;  // [NT,1024]
  gemm(X2t, 256, tW3t, tb3, X3t, 1024, NT, 1024, 256, 0);
  {
    dim3 g(Bn, (1024 + 255) / 256);
    segment_mean<<<g, 256, 0, stream>>>(X3t, idxt, offt, h_cat + 1024, 1024, 2080);
  }

  // ---- head ----
  head_fill<<<((Bn * 32) + 255) / 256, 256, 0, stream>>>(ref, h_cat, Bn);
  gemm(h_cat, 2080, fW1t, fb1, h1, 512, Bn, 512, 2080, 1);
  gemm(h1,     512, fW2t, fb2, h2,  32, Bn,  32,  512, 1);
  final_layer<<<(Bn + 255) / 256, 256, 0, stream>>>(h2, fW3, fb3, (float*)d_out, Bn, DT);
}
